// WaveNet_59957743452853
// MI455X (gfx1250) — matrix operand order repaired and checked by reading, not compiled, not run on hardware
//
#include <hip/hip_runtime.h>

typedef __attribute__((ext_vector_type(16))) _Float16 v16h;
typedef __attribute__((ext_vector_type(8)))  _Float16 v8h;
typedef __attribute__((ext_vector_type(8)))  float    v8f;

#define BB 4
#define TT 8192
#define MTOT (BB * TT)
#define CC 64
#define SS 256
#define VV 256
#define NLAYER 30

__device__ __forceinline__ v8f wmma_f16(v16h a, v16h b, v8f c) {
  // 8 args: (neg_a, A, neg_b, B, c_mod, C, reuse_a, reuse_b)
  return __builtin_amdgcn_wmma_f32_16x16x32_f16(false, a, false, b, (short)0, c,
                                                false, false);
}

// ---------------------------------------------------------------------------
// A-fragment (16x32 f16) per CDNA5 7.12.2: lane holds row = lane&15, two
// contiguous 8-element K-runs per 32-chunk: [half*8, +8) and [16+half*8, +8).
// ---------------------------------------------------------------------------
__device__ __forceinline__ v16h afrag_f32(const float* __restrict__ row, int kk,
                                          int half) {
  const float4* p = (const float4*)(row + kk * 32 + half * 8);
  const float4* q = (const float4*)(row + kk * 32 + 16 + half * 8);
  float4 a0 = p[0], a1 = p[1], b0 = q[0], b1 = q[1];
  v16h r;
  r[0] = (_Float16)a0.x;  r[1] = (_Float16)a0.y;
  r[2] = (_Float16)a0.z;  r[3] = (_Float16)a0.w;
  r[4] = (_Float16)a1.x;  r[5] = (_Float16)a1.y;
  r[6] = (_Float16)a1.z;  r[7] = (_Float16)a1.w;
  r[8] = (_Float16)b0.x;  r[9] = (_Float16)b0.y;
  r[10] = (_Float16)b0.z; r[11] = (_Float16)b0.w;
  r[12] = (_Float16)b1.x; r[13] = (_Float16)b1.y;
  r[14] = (_Float16)b1.z; r[15] = (_Float16)b1.w;
  return r;
}

// Same pattern from an f16 row in LDS: two 16-byte ds_load_b128 per fragment.
__device__ __forceinline__ v16h afrag_lds(const _Float16* __restrict__ row,
                                          int kk, int half) {
  v8h lo = *(const v8h*)(row + kk * 32 + half * 8);
  v8h hi = *(const v8h*)(row + kk * 32 + 16 + half * 8);
  v16h r;
#pragma unroll
  for (int i = 0; i < 8; ++i) { r[i] = lo[i]; r[8 + i] = hi[i]; }
  return r;
}

// ---------------------------------------------------------------------------
// Swizzled B-fragment storage: [frag][lane][16 halfs], 32B per lane,
// contiguous 1KB per fragment -> one v16h (2x b128) coalesced load per lane.
// ---------------------------------------------------------------------------
__device__ __forceinline__ v16h bfrag_sw(const _Float16* __restrict__ base,
                                         int frag, int lane) {
  return *(const v16h*)(base + ((size_t)frag * 32 + lane) * 16);
}

// One-time prep: swizzle [nmat][K][N] f32 weights into fragment-major f16.
// Thread u handles one (mat, frag, lane) and writes 16 contiguous halfs.
__global__ void swizzle_weights(const float* __restrict__ src,
                                _Float16* __restrict__ dst, int K, int N,
                                int nmat) {
  int u = blockIdx.x * blockDim.x + threadIdx.x;
  int nt = N / 16;
  int nfrag = (K / 32) * nt;
  if (u >= nmat * nfrag * 32) return;
  int lane = u & 31;
  int frag = (u >> 5) % nfrag;
  int mat = (u >> 5) / nfrag;
  int kk = frag / nt, j = frag % nt;
  int col = j * 16 + (lane & 15);
  int k0 = kk * 32 + ((lane >> 4) << 3);
  const float* s = src + (size_t)mat * K * N;
  v16h v;
#pragma unroll
  for (int i = 0; i < 16; ++i) v[i] = (_Float16)s[(size_t)(k0 + i + ((i >> 3) << 3)) * N + col];
  *(v16h*)(dst + (((size_t)mat * nfrag + frag) * 32 + lane) * 16) = v;
}

// ---------------------------------------------------------------------------
// Embedding gather: x[m] = embed[shifted_label(m)]
// ---------------------------------------------------------------------------
__global__ void wavenet_embed(const int* __restrict__ wf,
                              const float* __restrict__ embed,
                              float* __restrict__ x) {
  int m = blockIdx.x * blockDim.x + threadIdx.x;
  if (m >= MTOT) return;
  int t = m & (TT - 1);
  int idx = (t == 0) ? 128 : wf[m - 1];
  const float4* src = (const float4*)(embed + (size_t)idx * CC);
  float4* dst = (float4*)(x + (size_t)m * CC);
#pragma unroll
  for (int i = 0; i < CC / 4; ++i) dst[i] = src[i];
}

// ---------------------------------------------------------------------------
// One WaveNet layer: dilated conv (k=2) + gate + residual + skip accumulate.
// 256 threads = 8 waves, each wave owns 16 rows of M = B*T.
// ---------------------------------------------------------------------------
__global__ __launch_bounds__(256, 1) void wavenet_layer(
    const float* __restrict__ xin, float* __restrict__ xout,
    float* __restrict__ skip, const _Float16* __restrict__ w0s,
    const _Float16* __restrict__ w1s, const float* __restrict__ bc,
    const _Float16* __restrict__ rs, const float* __restrict__ br,
    const _Float16* __restrict__ sks, const float* __restrict__ bsk, int dil) {
  __shared__ _Float16 sG[8 * 16 * CC];  // per-wave gate tile [16][64] f16

  const int tid = threadIdx.x;
  const int wave = tid >> 5;
  const int lane = tid & 31;
  const int half = lane >> 4;
  const int arow = lane & 15;
  const int m0 = (blockIdx.x * 8 + wave) * 16;
  const int rowg = m0 + arow;
  const int t = rowg & (TT - 1);
  const bool has = (t >= dil);  // causal left-pad with zeros (per batch)

  const float* xr = xin + (size_t)rowg * CC;
  const float* xs = xin + (size_t)(rowg - dil) * CC;

  // A fragments: current x and dilated-shifted x, K = 64 -> 2 k-steps
  v16h ax[2], ash[2];
#pragma unroll
  for (int kk = 0; kk < 2; ++kk) {
    ax[kk] = afrag_f32(xr, kk, half);
    v16h z = {};
    ash[kk] = z;
    if (has) ash[kk] = afrag_f32(xs, kk, half);
  }

  // h = x_shift @ W0 + x @ W1 + b  -> 8 N-tiles of 16x16 f32
  v8f h[8];
#pragma unroll
  for (int j = 0; j < 8; ++j) {
    v8f acc = {};
    acc = wmma_f16(ash[0], bfrag_sw(w0s, 0 * 8 + j, lane), acc);
    acc = wmma_f16(ash[1], bfrag_sw(w0s, 1 * 8 + j, lane), acc);
    acc = wmma_f16(ax[0], bfrag_sw(w1s, 0 * 8 + j, lane), acc);
    acc = wmma_f16(ax[1], bfrag_sw(w1s, 1 * 8 + j, lane), acc);
    float bias = bc[j * 16 + arow];
#pragma unroll
    for (int g = 0; g < 8; ++g) acc[g] += bias;
    h[j] = acc;
  }

  // gate = tanh(h[:,:64]) * sigmoid(h[:,64:]) ; write to LDS in row-major f16
  // (C/D layout: element (M = g + 8*half, N = j*16 + arow))
  _Float16* gme = sG + wave * 16 * CC;
#pragma unroll
  for (int j = 0; j < 4; ++j) {
#pragma unroll
    for (int g = 0; g < 8; ++g) {
      float th = tanhf(h[j][g]);
      float sg = 1.0f / (1.0f + expf(-h[j + 4][g]));
      gme[(g + 8 * half) * CC + j * 16 + arow] = (_Float16)(th * sg);
    }
  }

  // Re-materialize gate as A fragments (layout transpose via LDS, b128 loads)
  v16h ga[2];
#pragma unroll
  for (int kk = 0; kk < 2; ++kk) ga[kk] = afrag_lds(gme + arow * CC, kk, half);

  // residual: x_out = x_in + gate @ res_w + res_b
#pragma unroll
  for (int j = 0; j < 4; ++j) {
    v8f acc = {};
    acc = wmma_f16(ga[0], bfrag_sw(rs, 0 * 4 + j, lane), acc);
    acc = wmma_f16(ga[1], bfrag_sw(rs, 1 * 4 + j, lane), acc);
    int col = j * 16 + arow;
    float bias = br[col];
#pragma unroll
    for (int g = 0; g < 8; ++g) {
      int row = m0 + g + 8 * half;
      xout[(size_t)row * CC + col] = xin[(size_t)row * CC + col] + acc[g] + bias;
    }
  }

  // skip accumulation: skip += gate @ skip_w + skip_b   (N = 256)
#pragma unroll
  for (int j = 0; j < 16; ++j) {
    v8f acc = {};
    acc = wmma_f16(ga[0], bfrag_sw(sks, 0 * 16 + j, lane), acc);
    acc = wmma_f16(ga[1], bfrag_sw(sks, 1 * 16 + j, lane), acc);
    int col = j * 16 + arow;
    float bias = bsk[col];
#pragma unroll
    for (int g = 0; g < 8; ++g) {
      int row = m0 + g + 8 * half;
      size_t idx = (size_t)row * SS + col;
      skip[idx] = skip[idx] + acc[g] + bias;
    }
  }
}

// ---------------------------------------------------------------------------
// Head: relu(skip @ w0 + b0) @ w1 + b1 -> log_softmax -> masked CE accumulate.
// 64 threads = 2 waves, each wave owns 16 rows.
// ---------------------------------------------------------------------------
__global__ __launch_bounds__(64, 1) void wavenet_head(
    const float* __restrict__ skip, const _Float16* __restrict__ w0s,
    const float* __restrict__ b0, const _Float16* __restrict__ w1s,
    const float* __restrict__ b1, const int* __restrict__ wf,
    const int* __restrict__ lens, float* __restrict__ accum) {
  __shared__ _Float16 sHid[2 * 16 * VV];  // hid0 as f16, per wave
  __shared__ float sLog[2 * 16 * VV];     // logits f32, per wave

  const int tid = threadIdx.x;
  const int wave = tid >> 5;
  const int lane = tid & 31;
  const int half = lane >> 4;
  const int arow = lane & 15;
  const int m0 = (blockIdx.x * 2 + wave) * 16;
  const int rowg = m0 + arow;

  // A fragments of skip_sum row, K = 256 -> 8 k-steps
  const float* sr = skip + (size_t)rowg * SS;
  v16h aS[8];
#pragma unroll
  for (int kk = 0; kk < 8; ++kk) aS[kk] = afrag_f32(sr, kk, half);

  // hid0 = relu(skip @ w0 + b0), stored f16 in LDS row-major
  _Float16* hme = sHid + wave * 16 * VV;
#pragma unroll
  for (int j = 0; j < 16; ++j) {
    v8f acc = {};
#pragma unroll
    for (int kk = 0; kk < 8; ++kk)
      acc = wmma_f16(aS[kk], bfrag_sw(w0s, kk * 16 + j, lane), acc);
    int col = j * 16 + arow;
    float bias = b0[col];
#pragma unroll
    for (int g = 0; g < 8; ++g) {
      float v = acc[g] + bias;
      hme[(g + 8 * half) * VV + col] = (_Float16)(v > 0.0f ? v : 0.0f);
    }
  }

  v16h aH[8];
#pragma unroll
  for (int kk = 0; kk < 8; ++kk) aH[kk] = afrag_lds(hme + arow * VV, kk, half);

  // logits = hid0 @ w1 + b1, stored f32 in LDS
  float* lme = sLog + wave * 16 * VV;
#pragma unroll
  for (int j = 0; j < 16; ++j) {
    v8f acc = {};
#pragma unroll
    for (int kk = 0; kk < 8; ++kk)
      acc = wmma_f16(aH[kk], bfrag_sw(w1s, kk * 16 + j, lane), acc);
    int col = j * 16 + arow;
    float bias = b1[col];
#pragma unroll
    for (int g = 0; g < 8; ++g) lme[(g + 8 * half) * VV + col] = acc[g] + bias;
  }

  // log-softmax + CE: lanes (r, r+16) cooperate on row r, 128 cols each
  const float* myrow = lme + arow * VV + half * 128;
  float mx = -3.4e38f;
  for (int c = 0; c < 128; ++c) mx = fmaxf(mx, myrow[c]);
  mx = fmaxf(mx, __shfl_xor(mx, 16, 32));
  float sum = 0.0f;
  for (int c = 0; c < 128; ++c) sum += expf(myrow[c] - mx);
  sum += __shfl_xor(sum, 16, 32);
  float lse = mx + logf(sum);

  if (half == 0) {
    int m = m0 + arow;
    int b = m >> 13;       // m / T
    int t = m & (TT - 1);  // m % T
    int label = wf[m];
    float ce = lse - lme[arow * VV + label];
    float msk = (t < lens[b]) ? 1.0f : 0.0f;
    atomicAdd(&accum[0], ce * msk);
    atomicAdd(&accum[1], msk);
  }
}

__global__ void wavenet_finalize(const float* __restrict__ accum,
                                 float* __restrict__ out) {
  out[0] = accum[0] / fmaxf(accum[1], 1.0f);
}

// ---------------------------------------------------------------------------
extern "C" void kernel_launch(void* const* d_in, const int* in_sizes, int n_in,
                              void* d_out, int out_size, void* d_ws,
                              size_t ws_size, hipStream_t stream) {
  (void)in_sizes; (void)n_in; (void)out_size; (void)ws_size;
  const int* wf = (const int*)d_in[0];
  const int* lens = (const int*)d_in[1];
  const float* embed = (const float*)d_in[2];
  const float* conv_w = (const float*)d_in[3];
  const float* conv_b = (const float*)d_in[4];
  const float* res_w = (const float*)d_in[5];
  const float* res_b = (const float*)d_in[6];
  const float* skip_w = (const float*)d_in[7];
  const float* skip_b = (const float*)d_in[8];
  const float* w0 = (const float*)d_in[9];
  const float* b0 = (const float*)d_in[10];
  const float* w1 = (const float*)d_in[11];
  const float* b1 = (const float*)d_in[12];

  char* ws = (char*)d_ws;
  const size_t xbytes = (size_t)MTOT * CC * sizeof(float);  // 8 MB
  const size_t sbytes = (size_t)MTOT * SS * sizeof(float);  // 32 MB
  float* x0 = (float*)ws;
  float* x1 = (float*)(ws + xbytes);
  float* skip = (float*)(ws + 2 * xbytes);
  float* accum = (float*)(ws + 2 * xbytes + sbytes);
  char* wsw = ws + 2 * xbytes + sbytes + 256;  // swizzled f16 weights
  _Float16* convS = (_Float16*)wsw;            // 60 mats of 64x128
  _Float16* resS = convS + (size_t)60 * CC * 2 * CC;   // 30 mats of 64x64
  _Float16* skipS = resS + (size_t)30 * CC * CC;       // 30 mats of 64x256
  _Float16* w0S = skipS + (size_t)30 * CC * SS;        // 1 mat of 256x256
  _Float16* w1S = w0S + (size_t)SS * VV;               // 1 mat of 256x256

  // One-time weight swizzles (cheap; graph-capture safe)
  {
    int n;
    n = 60 * (CC / 32) * (2 * CC / 16) * 32;
    swizzle_weights<<<(n + 255) / 256, 256, 0, stream>>>(conv_w, convS, CC,
                                                         2 * CC, 60);
    n = 30 * (CC / 32) * (CC / 16) * 32;
    swizzle_weights<<<(n + 255) / 256, 256, 0, stream>>>(res_w, resS, CC, CC,
                                                         30);
    n = 30 * (CC / 32) * (SS / 16) * 32;
    swizzle_weights<<<(n + 255) / 256, 256, 0, stream>>>(skip_w, skipS, CC, SS,
                                                         30);
    n = (SS / 32) * (VV / 16) * 32;
    swizzle_weights<<<(n + 255) / 256, 256, 0, stream>>>(w0, w0S, SS, VV, 1);
    swizzle_weights<<<(n + 255) / 256, 256, 0, stream>>>(w1, w1S, VV, VV, 1);
  }

  hipMemsetAsync(skip, 0, sbytes, stream);
  hipMemsetAsync(accum, 0, 2 * sizeof(float), stream);

  wavenet_embed<<<MTOT / 256, 256, 0, stream>>>(wf, embed, x0);

  static const int dil[NLAYER] = {1, 2, 4, 8, 16, 32, 64, 128, 256, 512,
                                  1, 2, 4, 8, 16, 32, 64, 128, 256, 512,
                                  1, 2, 4, 8, 16, 32, 64, 128, 256, 512};
  float* xin = x0;
  float* xout = x1;
  for (int l = 0; l < NLAYER; ++l) {
    wavenet_layer<<<MTOT / 128, 256, 0, stream>>>(
        xin, xout, skip, convS + (size_t)(2 * l) * CC * 2 * CC,
        convS + (size_t)(2 * l + 1) * CC * 2 * CC,
        conv_b + (size_t)l * 2 * CC, resS + (size_t)l * CC * CC,
        res_b + (size_t)l * CC, skipS + (size_t)l * CC * SS,
        skip_b + (size_t)l * SS, dil[l]);
    float* tmp = xin; xin = xout; xout = tmp;
  }

  wavenet_head<<<MTOT / 32, 64, 0, stream>>>(skip, w0S, b0, w1S, b1, wf, lens,
                                             accum);
  wavenet_finalize<<<1, 1, 0, stream>>>(accum, (float*)d_out);
}
